// WhaleAttention_17291538333976
// MI455X (gfx1250) — matrix operand order repaired and checked by reading, not compiled, not run on hardware
//
#include <hip/hip_runtime.h>

typedef __bf16 bf16;
typedef __attribute__((ext_vector_type(16))) __bf16 v16bf;
typedef __attribute__((ext_vector_type(8)))  float  v8f;
typedef __attribute__((ext_vector_type(4)))  float  v4f;
typedef __attribute__((ext_vector_type(4)))  unsigned int v4u;

#define WMMA_BF16(a, b, c) \
  __builtin_amdgcn_wmma_f32_16x16x32_bf16(false, (a), false, (b), (short)0, (c), false, false)

// ---- gfx1250 async global->LDS path (ASYNCcnt), guarded -------------------
#if defined(__has_builtin)
#if __has_builtin(__builtin_amdgcn_global_load_async_to_lds_b128) && \
    __has_builtin(__builtin_amdgcn_s_wait_asynccnt)
#define HAVE_ASYNC_LDS 1
#endif
#endif
#ifndef HAVE_ASYNC_LDS
#define HAVE_ASYNC_LDS 0
#endif

#if HAVE_ASYNC_LDS
typedef int v4i __attribute__((vector_size(16)));
typedef __attribute__((address_space(1))) v4i* as1_v4i;
typedef __attribute__((address_space(3))) v4i* as3_v4i;
__device__ __forceinline__ void async_cp16(const void* g, void* l) {
  __builtin_amdgcn_global_load_async_to_lds_b128(
      (as1_v4i)(v4i*)const_cast<void*>(g), (as3_v4i)(v4i*)l, 0, 0);
}
__device__ __forceinline__ void async_wait0() { __builtin_amdgcn_s_wait_asynccnt(0); }
#endif

constexpr int SA = 40;   // proj LDS stride (halves): 20-bank step, conflict-free
#define SQT 72           // attn bf16 tile stride (halves)
#define SST 65           // attn f32 score tile stride (floats)

// ---------------------------------------------------------------------------
// WMMA fragment loads from LDS (bf16, 16x16x32), wave32 layouts (ISA 7.12.2).
// ---------------------------------------------------------------------------
__device__ __forceinline__ v16bf lds_load_a(const bf16* base, int lda) {
  const int lane = threadIdx.x & 31;
  const int row  = lane & 15;
  const int kb   = (lane >> 4) << 3;
  const bf16* p  = base + row * lda + kb;
  v16bf r;
#pragma unroll
  for (int i = 0; i < 8; ++i) { r[i] = p[i]; r[i + 8] = p[i + 16]; }
  return r;
}

__device__ __forceinline__ v16bf lds_load_b(const bf16* base, int ldb) {
  const int lane = threadIdx.x & 31;
  const int col  = lane & 15;
  const int kb   = (lane >> 4) << 3;
  const bf16* p  = base + col * ldb + kb;
  v16bf r;
#pragma unroll
  for (int i = 0; i < 8; ++i) { r[i] = p[i]; r[i + 8] = p[i + 16]; }
  return r;
}

__device__ __forceinline__ void gemm8(const bf16* Ab, const bf16* Wb,
                                      int wm, int wn, v8f acc[2][4]) {
  v16bf a0 = lds_load_a(Ab + (wm + 0)  * SA, SA);
  v16bf a1 = lds_load_a(Ab + (wm + 16) * SA, SA);
  v16bf b0 = lds_load_b(Wb + (wn + 0)  * SA, SA);
  v16bf b1 = lds_load_b(Wb + (wn + 16) * SA, SA);
  v16bf b2 = lds_load_b(Wb + (wn + 32) * SA, SA);
  v16bf b3 = lds_load_b(Wb + (wn + 48) * SA, SA);
  acc[0][0] = WMMA_BF16(a0, b0, acc[0][0]);
  acc[0][1] = WMMA_BF16(a0, b1, acc[0][1]);
  acc[0][2] = WMMA_BF16(a0, b2, acc[0][2]);
  acc[0][3] = WMMA_BF16(a0, b3, acc[0][3]);
  acc[1][0] = WMMA_BF16(a1, b0, acc[1][0]);
  acc[1][1] = WMMA_BF16(a1, b1, acc[1][1]);
  acc[1][2] = WMMA_BF16(a1, b2, acc[1][2]);
  acc[1][3] = WMMA_BF16(a1, b3, acc[1][3]);
}

// ---------------------------------------------------------------------------
// Projection GEMM: Y = X(f32) @ W^T(f32) [+ bias] -> bf16.   K = N = 1024.
// MODE 0: out0=y+bias+pos_u, out1=y+bias+pos_v (Q); 1: y+bias (K,V); 2: y (P)
// BM=128, BN=128, BK=32, 256 thr, double-buffered software pipeline.
// ---------------------------------------------------------------------------
template <int MODE>
__global__ __launch_bounds__(256)
void proj_f32_bf16(const float* __restrict__ A, const float* __restrict__ W,
                   const float* __restrict__ bias,
                   const float* __restrict__ posu, const float* __restrict__ posv,
                   bf16* __restrict__ out0, bf16* __restrict__ out1) {
  constexpr int Kd = 1024, Nd = 1024;
  __shared__ bf16 As[2][128 * SA];
  __shared__ bf16 Ws[2][128 * SA];
  const int t = threadIdx.x, lane = t & 31, wave = t >> 5;
  const int m0 = blockIdx.x * 128, n0 = blockIdx.y * 128;
  const int wm = (wave >> 1) * 32, wn = (wave & 1) * 64;
  const int sr = t >> 1, sc = (t & 1) * 16;
  const float* gA = A + (size_t)(m0 + sr) * Kd + sc;
  const float* gW = W + (size_t)(n0 + sr) * Kd + sc;

  v4f ra[4], rw[4];
  auto gload = [&](int k0) {
#pragma unroll
    for (int q = 0; q < 4; ++q) {
      ra[q] = *(const v4f*)(gA + k0 + q * 4);
      rw[q] = *(const v4f*)(gW + k0 + q * 4);
    }
  };
  auto lstore = [&](int buf) {
#pragma unroll
    for (int q = 0; q < 4; ++q)
#pragma unroll
      for (int e = 0; e < 4; ++e) {
        As[buf][sr * SA + sc + q * 4 + e] = (bf16)ra[q][e];
        Ws[buf][sr * SA + sc + q * 4 + e] = (bf16)rw[q][e];
      }
  };

  v8f acc[2][4] = {};
  gload(0);
  lstore(0);
  __syncthreads();
  int cur = 0;
  for (int k0 = 0; k0 < Kd; k0 += 32) {
    const bool nxt = (k0 + 32) < Kd;
    if (nxt) gload(k0 + 32);
    gemm8(As[cur], Ws[cur], wm, wn, acc);
    if (nxt) lstore(cur ^ 1);
    __syncthreads();
    cur ^= 1;
  }

  // epilogue: constants folded into store offsets (Nd compile-time)
  const int colL = lane & 15, rowHi = (lane >> 4) << 3;
  float bcol[4], pu[4], pv[4];
#pragma unroll
  for (int j = 0; j < 4; ++j) {
    int col = n0 + wn + j * 16 + colL;
    bcol[j] = (MODE != 2) ? bias[col] : 0.0f;
    if (MODE == 0) { pu[j] = posu[col]; pv[j] = posv[col]; }
  }
  const size_t base = (size_t)(m0 + wm + rowHi) * Nd + n0 + wn + colL;
  bf16* p0 = out0 + base;
  bf16* p1 = (MODE == 0) ? (out1 + base) : nullptr;
#pragma unroll
  for (int i = 0; i < 2; ++i)
#pragma unroll
    for (int j = 0; j < 4; ++j)
#pragma unroll
      for (int r = 0; r < 8; ++r) {
        const int off = (i * 16 + r) * Nd + j * 16;
        float y = acc[i][j][r] + bcol[j];
        if (MODE == 0) {
          p0[off] = (bf16)(y + pu[j]);
          p1[off] = (bf16)(y + pv[j]);
        } else {
          p0[off] = (bf16)y;
        }
      }
}

// ---------------------------------------------------------------------------
// Output projection: out(f32) = A(bf16) @ W^T(f32) + bias.   K = N = 1024.
// A tile streamed with async global->LDS (double-buffered); W reg-pipelined.
// ---------------------------------------------------------------------------
__global__ __launch_bounds__(256)
void proj_bf16_f32(const bf16* __restrict__ A, const float* __restrict__ W,
                   const float* __restrict__ bias, float* __restrict__ out) {
  constexpr int Kd = 1024, Nd = 1024;
  __shared__ bf16 As[2][128 * SA];
  __shared__ bf16 Ws[2][128 * SA];
  const int t = threadIdx.x, lane = t & 31, wave = t >> 5;
  const int m0 = blockIdx.x * 128, n0 = blockIdx.y * 128;
  const int wm = (wave >> 1) * 32, wn = (wave & 1) * 64;
  const int sr = t >> 1, sc = (t & 1) * 16;
  const float* gW = W + (size_t)(n0 + sr) * Kd + sc;

  v4f rw[4];
  auto wload = [&](int k0) {
#pragma unroll
    for (int q = 0; q < 4; ++q) rw[q] = *(const v4f*)(gW + k0 + q * 4);
  };
  auto wstore = [&](int buf) {
#pragma unroll
    for (int q = 0; q < 4; ++q)
#pragma unroll
      for (int e = 0; e < 4; ++e) Ws[buf][sr * SA + sc + q * 4 + e] = (bf16)rw[q][e];
  };

#if HAVE_ASYNC_LDS
  const int ar0 = (t * 16) >> 6,          ac0 = (t * 16) & 63;
  const int ar1 = ((t * 16) + 4096) >> 6, ac1 = ((t * 16) + 4096) & 63;
  auto aastage = [&](int k0, int buf) {
    async_cp16((const char*)A + ((size_t)(m0 + ar0) * Kd + k0) * 2 + ac0,
               (char*)As[buf] + ar0 * SA * 2 + ac0);
    async_cp16((const char*)A + ((size_t)(m0 + ar1) * Kd + k0) * 2 + ac1,
               (char*)As[buf] + ar1 * SA * 2 + ac1);
  };
#else
  const bf16* gAb = A + (size_t)(m0 + sr) * Kd + sc;
  v4u raA[2];
  auto aload = [&](int k0) {
    raA[0] = *(const v4u*)(gAb + k0);
    raA[1] = *(const v4u*)(gAb + k0 + 8);
  };
  auto astore = [&](int buf) {
    v4u* d = (v4u*)(As[buf] + sr * SA + sc);
    d[0] = raA[0]; d[1] = raA[1];
  };
#endif

  v8f acc[2][4] = {};
#if HAVE_ASYNC_LDS
  aastage(0, 0);
#else
  aload(0); astore(0);
#endif
  wload(0); wstore(0);
#if HAVE_ASYNC_LDS
  async_wait0();
#endif
  __syncthreads();

  int cur = 0;
  for (int k0 = 0; k0 < Kd; k0 += 32) {
    const bool nxt = (k0 + 32) < Kd;
    if (nxt) {
#if HAVE_ASYNC_LDS
      aastage(k0 + 32, cur ^ 1);
#else
      aload(k0 + 32);
#endif
      wload(k0 + 32);
    }
    gemm8(As[cur], Ws[cur], wm, wn, acc);
    if (nxt) {
#if !HAVE_ASYNC_LDS
      astore(cur ^ 1);
#endif
      wstore(cur ^ 1);
#if HAVE_ASYNC_LDS
      async_wait0();
#endif
    }
    __syncthreads();
    cur ^= 1;
  }

  const int colL = lane & 15, rowHi = (lane >> 4) << 3;
  float bcol[4];
#pragma unroll
  for (int j = 0; j < 4; ++j) bcol[j] = bias[n0 + wn + j * 16 + colL];
  float* po = out + (size_t)(m0 + wm + rowHi) * Nd + n0 + wn + colL;
#pragma unroll
  for (int i = 0; i < 2; ++i)
#pragma unroll
    for (int j = 0; j < 4; ++j)
#pragma unroll
      for (int r = 0; r < 8; ++r)
        po[(i * 16 + r) * Nd + j * 16] = acc[i][j][r] + bcol[j];
}

// ---------------------------------------------------------------------------
// Flash attention with relative-position bias (N=2048, D=64, C=1024):
//   S = (Q_u K^T + Q_v P^T) * 0.125 ; P = softmax(S) ; O = P V
// Block = (batch, head, 128-query tile); 256 threads; double-buffered
// 64-key chunks: K/P via async-to-LDS, V register-staged + transposed.
// ---------------------------------------------------------------------------
#define ATTN_SMEM_BYTES (128*SQT*2*2 + 64*SQT*2*6 + 128*SQT*2 + 128*SST*4 + 3*128*4 + 256)

__global__ __launch_bounds__(256)
void attn_kernel(const bf16* __restrict__ qu, const bf16* __restrict__ qv,
                 const bf16* __restrict__ kk, const bf16* __restrict__ pp,
                 const bf16* __restrict__ vv, bf16* __restrict__ outO) {
  constexpr int Nn = 2048;
  constexpr float scale = 0.125f;
  extern __shared__ char smem[];
  bf16*  squ   = (bf16*)smem;                 // 128 x SQT
  bf16*  sqv   = squ + 128 * SQT;             // 128 x SQT
  bf16*  skb   = sqv + 128 * SQT;             // 2 x 64 x SQT (K chunks)
  bf16*  spb   = skb + 2 * 64 * SQT;          // 2 x 64 x SQT (P chunks)
  bf16*  svb   = spb + 2 * 64 * SQT;          // 2 x 64 x SQT (V^T chunks)
  bf16*  sprob = svb + 2 * 64 * SQT;          // 128 x SQT (probs)
  float* ss    = (float*)(sprob + 128 * SQT); // 128 x SST scores
  float* srm   = ss  + 128 * SST;
  float* srl   = srm + 128;
  float* salph = srl + 128;

  const int t = threadIdx.x, lane = t & 31, wave = t >> 5;
  const int n0 = blockIdx.x * 128;
  const int h  = blockIdx.y;
  const int b  = blockIdx.z;
  const size_t headOff  = (size_t)h * 64;
  const size_t batchOff = (size_t)b * Nn * 1024;

  // ---- V chunk: register stage + transpose store -------------------------
  const int vr = t >> 2, vc = (t & 3) * 16;
  const size_t gvoff = batchOff + headOff + (size_t)vr * 1024 + vc;
  v4u rvv[2];
  auto vload = [&](int j) {
    const v4u* p = (const v4u*)(vv + gvoff + (size_t)j * 1024);
    rvv[0] = p[0]; rvv[1] = p[1];
  };
  auto vstore = [&](bf16* dst) {
    unsigned short* d = (unsigned short*)dst;
    unsigned int dw[8] = {rvv[0][0], rvv[0][1], rvv[0][2], rvv[0][3],
                          rvv[1][0], rvv[1][1], rvv[1][2], rvv[1][3]};
#pragma unroll
    for (int e = 0; e < 8; ++e) {
      d[(vc + 2 * e + 0) * SQT + vr] = (unsigned short)(dw[e] & 0xffffu);
      d[(vc + 2 * e + 1) * SQT + vr] = (unsigned short)(dw[e] >> 16);
    }
  };

  // ---- K / P chunk staging ------------------------------------------------
#if HAVE_ASYNC_LDS
  auto kp_stage = [&](int j, bf16* dk, bf16* dp) {
#pragma unroll
    for (int q = 0; q < 2; ++q) {
      int o = t * 16 + q * 4096;
      int r = o >> 7, c = o & 127;
      size_t gb = (batchOff + (size_t)(j + r) * 1024 + headOff) * 2 + c;
      async_cp16((const char*)kk + gb, (char*)dk + (size_t)r * SQT * 2 + c);
      async_cp16((const char*)pp + gb, (char*)dp + (size_t)r * SQT * 2 + c);
    }
  };
#else
  v4u rkk[2], rpp[2];
  auto kp_load = [&](int j) {
    const v4u* pk = (const v4u*)(kk + gvoff + (size_t)j * 1024);
    const v4u* pq = (const v4u*)(pp + gvoff + (size_t)j * 1024);
    rkk[0] = pk[0]; rkk[1] = pk[1];
    rpp[0] = pq[0]; rpp[1] = pq[1];
  };
  auto kp_store = [&](bf16* dk, bf16* dp) {
    v4u* a = (v4u*)(dk + vr * SQT + vc);
    v4u* c2 = (v4u*)(dp + vr * SQT + vc);
    a[0] = rkk[0]; a[1] = rkk[1];
    c2[0] = rpp[0]; c2[1] = rpp[1];
  };
#endif

  // ---- prologue: Q tiles + chunk 0 ---------------------------------------
#if HAVE_ASYNC_LDS
#pragma unroll
  for (int q = 0; q < 4; ++q) {
    int o = t * 16 + q * 4096;
    int r = o >> 7, c = o & 127;
    size_t gb = (batchOff + (size_t)(n0 + r) * 1024 + headOff) * 2 + c;
    async_cp16((const char*)qu + gb, (char*)squ + (size_t)r * SQT * 2 + c);
    async_cp16((const char*)qv + gb, (char*)sqv + (size_t)r * SQT * 2 + c);
  }
  kp_stage(0, skb, spb);
#else
  {
    const int qr = t >> 1, qc = (t & 1) * 32;
    const bf16* gu = qu + batchOff + (size_t)(n0 + qr) * 1024 + headOff + qc;
    const bf16* gv = qv + batchOff + (size_t)(n0 + qr) * 1024 + headOff + qc;
#pragma unroll
    for (int q = 0; q < 4; ++q) {
      ((v4u*)(squ + qr * SQT + qc))[q] = ((const v4u*)gu)[q];
      ((v4u*)(sqv + qr * SQT + qc))[q] = ((const v4u*)gv)[q];
    }
  }
  kp_load(0); kp_store(skb, spb);
#endif
  vload(0); vstore(svb);
  if (t < 128) { srm[t] = -1e30f; srl[t] = 0.0f; }
#if HAVE_ASYNC_LDS
  async_wait0();
#endif
  __syncthreads();

  const int colL = lane & 15, rowHi = (lane >> 4) << 3;
  v8f oacc[4] = {};
  int cur = 0;

  for (int j = 0; j < Nn; j += 64) {
    const bool nxt = (j + 64) < Nn;
    bf16* skc = skb + cur * 64 * SQT;
    bf16* spc = spb + cur * 64 * SQT;
    bf16* svc = svb + cur * 64 * SQT;
    if (nxt) {
#if HAVE_ASYNC_LDS
      kp_stage(j + 64, skb + (cur ^ 1) * 64 * SQT, spb + (cur ^ 1) * 64 * SQT);
#else
      kp_load(j + 64);
#endif
      vload(j + 64);
    }

    // ---- S = (Q_u K^T + Q_v P^T) * scale ----
    v16bf aqu0 = lds_load_a(squ + wave * 16 * SQT + 0,  SQT);
    v16bf aqu1 = lds_load_a(squ + wave * 16 * SQT + 32, SQT);
    v16bf aqv0 = lds_load_a(sqv + wave * 16 * SQT + 0,  SQT);
    v16bf aqv1 = lds_load_a(sqv + wave * 16 * SQT + 32, SQT);
#pragma unroll
    for (int ct = 0; ct < 4; ++ct) {
      v8f s = {};
      v16bf bk0 = lds_load_b(skc + ct * 16 * SQT + 0,  SQT);
      v16bf bk1 = lds_load_b(skc + ct * 16 * SQT + 32, SQT);
      v16bf bp0 = lds_load_b(spc + ct * 16 * SQT + 0,  SQT);
      v16bf bp1 = lds_load_b(spc + ct * 16 * SQT + 32, SQT);
      s = WMMA_BF16(aqu0, bk0, s);
      s = WMMA_BF16(aqu1, bk1, s);
      s = WMMA_BF16(aqv0, bp0, s);
      s = WMMA_BF16(aqv1, bp1, s);
#pragma unroll
      for (int r = 0; r < 8; ++r)
        ss[(wave * 16 + r + rowHi) * SST + ct * 16 + colL] = s[r] * scale;
    }
    __syncthreads();

    // ---- online softmax: one thread per query row ----
    if (t < 128) {
      float m_old = srm[t];
      float cmax = -1e30f;
#pragma unroll 8
      for (int i = 0; i < 64; ++i) cmax = fmaxf(cmax, ss[t * SST + i]);
      float m_new = fmaxf(m_old, cmax);
      float alpha = __expf(m_old - m_new);
      float psum = 0.0f;
#pragma unroll 8
      for (int i = 0; i < 64; ++i) {
        float e = __expf(ss[t * SST + i] - m_new);
        sprob[t * SQT + i] = (bf16)e;
        psum += e;
      }
      srl[t] = srl[t] * alpha + psum;
      srm[t] = m_new;
      salph[t] = alpha;
    }
    __syncthreads();

    if (nxt) {
      vstore(svb + (cur ^ 1) * 64 * SQT);
#if !HAVE_ASYNC_LDS
      kp_store(skb + (cur ^ 1) * 64 * SQT, spb + (cur ^ 1) * 64 * SQT);
#endif
    }

    // ---- O = O*alpha + P @ V ----
    float arow[8];
#pragma unroll
    for (int r = 0; r < 8; ++r) arow[r] = salph[wave * 16 + r + rowHi];
    v16bf ap0 = lds_load_a(sprob + wave * 16 * SQT + 0,  SQT);
    v16bf ap1 = lds_load_a(sprob + wave * 16 * SQT + 32, SQT);
#pragma unroll
    for (int dt = 0; dt < 4; ++dt) {
#pragma unroll
      for (int r = 0; r < 8; ++r) oacc[dt][r] *= arow[r];
      v16bf bv0 = lds_load_b(svc + dt * 16 * SQT + 0,  SQT);
      v16bf bv1 = lds_load_b(svc + dt * 16 * SQT + 32, SQT);
      oacc[dt] = WMMA_BF16(ap0, bv0, oacc[dt]);
      oacc[dt] = WMMA_BF16(ap1, bv1, oacc[dt]);
    }
#if HAVE_ASYNC_LDS
    if (nxt) async_wait0();
#endif
    __syncthreads();
    cur ^= 1;
  }

  // ---- finalize: O /= l, write bf16 (B,N,C) ----
  float rl[8];
#pragma unroll
  for (int r = 0; r < 8; ++r) rl[r] = 1.0f / srl[wave * 16 + r + rowHi];
  bf16* po = outO + batchOff + (size_t)(n0 + wave * 16 + rowHi) * 1024 + headOff + colL;
#pragma unroll
  for (int dt = 0; dt < 4; ++dt)
#pragma unroll
    for (int r = 0; r < 8; ++r)
      po[r * 1024 + dt * 16] = (bf16)(oacc[dt][r] * rl[r]);
}

// ---------------------------------------------------------------------------
extern "C" void kernel_launch(void* const* d_in, const int* in_sizes, int n_in,
                              void* d_out, int out_size, void* d_ws, size_t ws_size,
                              hipStream_t stream) {
  const float* hidden = (const float*)d_in[0];
  const float* pos    = (const float*)d_in[1];
  const float* Wq     = (const float*)d_in[2];
  const float* bq     = (const float*)d_in[3];
  const float* Wk     = (const float*)d_in[4];
  const float* bk     = (const float*)d_in[5];
  const float* Wv     = (const float*)d_in[6];
  const float* bv     = (const float*)d_in[7];
  const float* Wpos   = (const float*)d_in[8];
  const float* pbu    = (const float*)d_in[9];
  const float* pbv    = (const float*)d_in[10];
  const float* Wo     = (const float*)d_in[11];
  const float* bo     = (const float*)d_in[12];

  const int Bsz = 2, Nn = 2048, Cc = 1024;
  const int M = Bsz * Nn;                 // 4096
  const size_t tile = (size_t)M * Cc;

  bf16* qu = (bf16*)d_ws;
  bf16* qv = qu + tile;
  bf16* kb = qv + tile;
  bf16* vb = kb + tile;
  bf16* pb = vb + tile;
  bf16* ao = pb + tile;                   // 48 MB total workspace

  dim3 blk(256);
  dim3 gP(M / 128, Cc / 128);             // 32 x 8

  proj_f32_bf16<0><<<gP, blk, 0, stream>>>(hidden, Wq, bq, pbu, pbv, qu, qv);
  proj_f32_bf16<1><<<gP, blk, 0, stream>>>(hidden, Wk, bk, nullptr, nullptr, kb, nullptr);
  proj_f32_bf16<1><<<gP, blk, 0, stream>>>(hidden, Wv, bv, nullptr, nullptr, vb, nullptr);
  proj_f32_bf16<2><<<gP, blk, 0, stream>>>(pos,    Wpos, nullptr, nullptr, nullptr, pb, nullptr);

  dim3 gA(Nn / 128, 16 /*H*/, Bsz);
  attn_kernel<<<gA, blk, ATTN_SMEM_BYTES, stream>>>(qu, qv, kb, pb, vb, ao);

  proj_bf16_f32<<<gP, blk, 0, stream>>>(ao, Wo, bo, (float*)d_out);
}
